// Interact_12532714569788
// MI455X (gfx1250) — compile-verified, matrix operand order checked by reading, not run on hardware
//
#include <hip/hip_runtime.h>
#include <math.h>

// Problem dims (match reference)
#define Bd   32
#define Td   96
#define Hd   384
#define Ed   384
#define NSPK 8
#define H2   (2*Hd)   // 768
#define H3   (3*Hd)   // 1152
#define H4   (4*Hd)   // 1536
#define H8   (8*Hd)   // 3072

typedef __attribute__((ext_vector_type(16))) _Float16 v16h;
typedef __attribute__((ext_vector_type(8)))  _Float16 v8h;
typedef __attribute__((ext_vector_type(8)))  float    v8f;
typedef _Float16 f16;

__device__ __forceinline__ float sigf(float x) { return 1.0f / (1.0f + __expf(-x)); }

// ---------------------------------------------------------------------------
// WMMA GEMM:  D[M,N] = A[M,K] @ W[N,K]^T (+ bias[N]) (+ addend[M,N])
// A and W are pre-converted f16.  One wave per 16x16 tile, K templated so the
// loop fully unrolls into a back-to-back v_wmma stream with b128 loads.
//
// Fragment striping (CDNA5 ISA 7.12.2, 16-bit A 16x32):
//   lane&15 = M row (A) / N col (B); v16h elements [8g..8g+7] hold
//   k = g*16 + (lane>>4)*8 + 0..7  ->  two contiguous 16B chunks per fragment.
// ---------------------------------------------------------------------------
template <int K>
__global__ __launch_bounds__(32) void wmma_gemm(
    const f16* __restrict__ A, int lda,
    const f16* __restrict__ W, int ldw,
    const float* __restrict__ bias,
    const float* __restrict__ addend, int ldadd,
    float* __restrict__ D, int ldd,
    f16* __restrict__ D16, int ldd16)
{
  const int tileN = blockIdx.x, tileM = blockIdx.y;
  const int lane = threadIdx.x & 31;
  const int hi8  = (lane >> 4) << 3;
  const int r16  = lane & 15;
  const f16* Ar = A + (size_t)(tileM * 16 + r16) * lda;
  const f16* Wr = W + (size_t)(tileN * 16 + r16) * ldw;

  v8f acc = {};
#pragma unroll
  for (int k0 = 0; k0 < K; k0 += 32) {
    const v8h a0 = *(const v8h*)(Ar + k0 + hi8);
    const v8h a1 = *(const v8h*)(Ar + k0 + 16 + hi8);
    const v8h b0 = *(const v8h*)(Wr + k0 + hi8);
    const v8h b1 = *(const v8h*)(Wr + k0 + 16 + hi8);
    v16h a, b;
#pragma unroll
    for (int i = 0; i < 8; ++i) {
      a[i] = a0[i]; a[8 + i] = a1[i];
      b[i] = b0[i]; b[8 + i] = b1[i];
    }
    acc = __builtin_amdgcn_wmma_f32_16x16x32_f16(
        false, a, false, b, (short)0, acc, false, false);
  }

  // C/D layout: lane&15 = N, VGPR r -> M = r + 8*(lane>>4)
  const int n  = tileN * 16 + r16;
  const float bv = bias ? bias[n] : 0.0f;
#pragma unroll
  for (int r = 0; r < 8; ++r) {
    const int m = tileM * 16 + (hi8 ? 8 : 0) + r;
    float val = acc[r] + bv;
    if (addend) val += addend[(size_t)m * ldadd + n];
    D[(size_t)m * ldd + n] = val;
    if (D16) D16[(size_t)m * ldd16 + n] = (f16)val;
  }
}

static inline void gemm(hipStream_t s, const f16* A, int lda,
                        const f16* W, int ldw, const float* bias,
                        const float* addend, int ldadd,
                        float* D, int ldd, f16* D16, int ldd16,
                        int M, int N, int K) {
  dim3 grid(N / 16, M / 16);
  if (K == 384)
    wmma_gemm<384><<<grid, 32, 0, s>>>(A, lda, W, ldw, bias, addend, ldadd, D, ldd, D16, ldd16);
  else
    wmma_gemm<768><<<grid, 32, 0, s>>>(A, lda, W, ldw, bias, addend, ldadd, D, ldd, D16, ldd16);
}

// ------------------------------ init kernels ------------------------------
__global__ void k_f32tof16(const float* __restrict__ src, f16* __restrict__ dst, int n) {
  int i = blockIdx.x * blockDim.x + threadIdx.x;
  if (i < n) dst[i] = (f16)src[i];
}

// embedding gather straight to f16 (GEMM A operand)
__global__ void k_embed(const float* __restrict__ tab, const int* __restrict__ ids,
                        f16* __restrict__ out, int total) {
  int i = blockIdx.x * blockDim.x + threadIdx.x;
  if (i >= total) return;
  out[i] = (f16)tab[(size_t)ids[i / Ed] * Ed + (i % Ed)];
}

__global__ void k_addvec(const float* __restrict__ a, const float* __restrict__ b,
                         float* __restrict__ o, int n) {
  int i = blockIdx.x * blockDim.x + threadIdx.x;
  if (i < n) o[i] = a[i] + b[i];
}

// gbias[n] = bih_g[n] + bhh_g[n] + dot(Whh_g[n,:], d_h)   (constant per call)
__global__ void k_gbias(const float* __restrict__ Whh_g, const float* __restrict__ d_h,
                        const float* __restrict__ bih, const float* __restrict__ bhh,
                        float* __restrict__ o) {
  int n = blockIdx.x * blockDim.x + threadIdx.x;
  if (n >= H4) return;
  float s = bih[n] + bhh[n];
  const float* w = Whh_g + (size_t)n * Hd;
  for (int k = 0; k < Hd; ++k) s += w[k] * d_h[k];
  o[n] = s;
}

// vinit[n] = dot(bk, Wv[n,:]) + bv[n]
__global__ void k_vinit(const float* __restrict__ Wv, const float* __restrict__ bk,
                        const float* __restrict__ bv, float* __restrict__ o) {
  int n = blockIdx.x * blockDim.x + threadIdx.x;
  if (n >= Hd) return;
  float s = bv[n];
  const float* w = Wv + (size_t)n * Hd;
  for (int k = 0; k < Hd; ++k) s += w[k] * bk[k];
  o[n] = s;
}

__global__ void k_initKV(float* __restrict__ Kb, float* __restrict__ Vb,
                         const float* __restrict__ bk, const float* __restrict__ vinit,
                         int total) {
  int i = blockIdx.x * blockDim.x + threadIdx.x;
  if (i >= total) return;
  Kb[i] = bk[i % Hd];
  Vb[i] = vinit[i % Hd];
}

__global__ void k_initspk(float* __restrict__ spk, const float* __restrict__ h0, int total) {
  int i = blockIdx.x * blockDim.x + threadIdx.x;
  if (i < total) spk[i] = h0[i % H2];
}

// ------------------------------ phase 1 cell ------------------------------
__global__ void k_lstm_d(const float* __restrict__ gates, f16* __restrict__ h16,
                         float* __restrict__ c_d, float* __restrict__ dial,
                         f16* __restrict__ dial16, int t) {
  int i = blockIdx.x * blockDim.x + threadIdx.x;        // over B*H
  if (i >= Bd * Hd) return;
  int b = i / Hd, e = i % Hd;
  const float* g = gates + (size_t)b * H4;
  float ig = sigf(g[e]), fg = sigf(g[Hd + e]);
  float gg = tanhf(g[2 * Hd + e]), og = sigf(g[3 * Hd + e]);
  float c2 = fg * c_d[i] + ig * gg;
  c_d[i] = c2;
  float h2 = og * tanhf(c2);
  h16[i] = (f16)h2;                                     // next-step GEMM A
  dial[((size_t)b * Td + t) * Hd + e]   = h2;
  dial16[((size_t)b * Td + t) * Hd + e] = (f16)h2;      // q-GEMM A rows
}

// ------------------------------ attention ---------------------------------
__global__ void k_attn(const float* __restrict__ qb, const float* __restrict__ dial,
                       const float* __restrict__ Kb, const float* __restrict__ Vb,
                       const float* __restrict__ att_h, const int* __restrict__ chat_ids,
                       const int* __restrict__ spk_info, const float* __restrict__ spk_h,
                       f16* __restrict__ x16, f16* __restrict__ hsp16, int t) {
  __shared__ float sq[Hd];
  __shared__ float sc[Td];
  __shared__ float sinv;
  const int b = blockIdx.x, tid = threadIdx.x, NT = blockDim.x;
  for (int d = tid; d < Hd; d += NT) sq[d] = qb[b * Hd + d];
  __syncthreads();
  const float scale = 1.0f / sqrtf((float)Hd);
  for (int j = tid; j <= t; j += NT) {
    const float* kr = Kb + ((size_t)b * Td + j) * Hd;
    __builtin_prefetch(Vb + ((size_t)b * Td + j) * Hd, 0, 3);  // global_prefetch_b8
    float s = 0.f;
    for (int k = 0; k < Hd; ++k) s += kr[k] * sq[k];
    sc[j] = s * scale;
  }
  __syncthreads();
  if (tid == 0) {                      // softmax over <=96 scores: serial is fine
    float mx = -1e30f;
    for (int j = 0; j <= t; ++j) mx = fmaxf(mx, sc[j]);
    float sum = 0.f;
    for (int j = 0; j <= t; ++j) { float p = __expf(sc[j] - mx); sc[j] = p; sum += p; }
    sinv = 1.0f / sum;
  }
  __syncthreads();
  const float inv = sinv;
  for (int d = tid; d < Hd; d += NT) {
    float a = 0.f;
    for (int j = 0; j <= t; ++j) a += sc[j] * Vb[((size_t)b * Td + j) * Hd + d];
    a *= inv;
    // x_s = concat(t==0 ? attention_h : attn, u)  -> f16 GEMM A operand
    x16[(size_t)b * H2 + d]      = (f16)((t == 0) ? att_h[d] : a);
    x16[(size_t)b * H2 + Hd + d] = (f16)dial[((size_t)b * Td + t) * Hd + d];
  }
  const int s = spk_info[chat_ids[b] * Td + t];
  for (int d = tid; d < H2; d += NT)
    hsp16[(size_t)b * H2 + d] = (f16)spk_h[((size_t)b * NSPK + s) * H2 + d];
}

// ------------------------------ speaker LSTM cell -------------------------
__global__ void k_lstm_s(const float* __restrict__ gates, const float* __restrict__ dial,
                         const float* __restrict__ att_h, const int* __restrict__ chat_ids,
                         const int* __restrict__ spk_info, float* __restrict__ c_s,
                         float* __restrict__ spk_h, float* __restrict__ cg,
                         float* __restrict__ out_spop, f16* __restrict__ spop16, int t) {
  int i = blockIdx.x * blockDim.x + threadIdx.x;        // over B*2H
  if (i >= Bd * H2) return;
  int b = i / H2, e = i % H2;
  const float* g = gates + (size_t)b * H8;
  float ig = sigf(g[e]), fg = sigf(g[H2 + e]);
  float gg = tanhf(g[2 * H2 + e]), og = sigf(g[3 * H2 + e]);
  float cin = (t == 0) ? 0.0f : c_s[i];
  float c2 = fg * cin + ig * gg;
  c_s[i] = c2;
  float h = og * tanhf(c2);
  float au = (e < Hd) ? att_h[e] : dial[((size_t)b * Td + t) * Hd + (e - Hd)];
  float sp = h + au;                                    // spop = h_s + au
  out_spop[((size_t)b * Td + t) * H2 + e] = sp;
  spop16[(size_t)b * H2 + e] = (f16)sp;                 // gates_g GEMM A
  int s = spk_info[chat_ids[b] * Td + t];
  spk_h[((size_t)b * NSPK + s) * H2 + e] = h;
  if (e < Hd) cg[b * Hd + e] = c2 / (float)(t + 1);     // avg_cell[:H]
}

// ------------------------------ global LSTM cell --------------------------
__global__ void k_lstm_g(const float* __restrict__ gates, const float* __restrict__ cg,
                         float* __restrict__ out_op, f16* __restrict__ hg16, int t) {
  int i = blockIdx.x * blockDim.x + threadIdx.x;        // over B*H
  if (i >= Bd * Hd) return;
  int b = i / Hd, e = i % Hd;
  const float* g = gates + (size_t)b * H4;
  float ig = sigf(g[e]), fg = sigf(g[Hd + e]);
  float gg = tanhf(g[2 * Hd + e]), og = sigf(g[3 * Hd + e]);
  float c2 = fg * cg[i] + ig * gg;
  float hv = og * tanhf(c2);
  out_op[((size_t)b * Td + t) * Hd + e] = hv;
  hg16[b * Hd + e] = (f16)hv;                           // k-GEMM A
}

// ---------------------------------------------------------------------------
extern "C" void kernel_launch(void* const* d_in, const int* in_sizes, int n_in,
                              void* d_out, int out_size, void* d_ws, size_t ws_size,
                              hipStream_t stream) {
  (void)in_sizes; (void)n_in; (void)out_size; (void)ws_size;
  const float* emb_table = (const float*)d_in[0];
  const float* Wih_d = (const float*)d_in[1];
  const float* Whh_d = (const float*)d_in[2];
  const float* bih_d = (const float*)d_in[3];
  const float* bhh_d = (const float*)d_in[4];
  const float* Wih_s = (const float*)d_in[5];
  const float* Whh_s = (const float*)d_in[6];
  const float* bih_s = (const float*)d_in[7];
  const float* bhh_s = (const float*)d_in[8];
  const float* Wih_g = (const float*)d_in[9];
  const float* Whh_g = (const float*)d_in[10];
  const float* bih_g = (const float*)d_in[11];
  const float* bhh_g = (const float*)d_in[12];
  const float* Wq = (const float*)d_in[13];
  const float* bq = (const float*)d_in[14];
  const float* Wk = (const float*)d_in[15];
  const float* bk = (const float*)d_in[16];
  const float* Wv = (const float*)d_in[17];
  const float* bv = (const float*)d_in[18];
  const float* h_0   = (const float*)d_in[19];
  const float* d_h   = (const float*)d_in[20];
  const float* att_h = (const float*)d_in[21];
  const int* chat_ids = (const int*)d_in[22];
  const int* spk_info = (const int*)d_in[23];
  const int* inputs   = (const int*)d_in[26];

  float* out_op   = (float*)d_out;                      // [B,T,H]
  float* out_spop = out_op + (size_t)Bd * Td * Hd;      // [B,T,2H]

  // ---- workspace carve: f32 region then f16 region (all 64B aligned) ----
  float* w = (float*)d_ws;
  float* xg      = w; w += (size_t)Bd * Td * H4;
  float* gates_d = w; w += (size_t)Bd * H4;
  float* c_d     = w; w += (size_t)Bd * Hd;
  float* dial    = w; w += (size_t)Bd * Td * Hd;
  float* Kb      = w; w += (size_t)Bd * Td * Hd;
  float* Vb      = w; w += (size_t)Bd * Td * Hd;
  float* spk_h   = w; w += (size_t)Bd * NSPK * H2;
  float* c_s     = w; w += (size_t)Bd * H2;
  float* qb      = w; w += (size_t)Bd * Hd;
  float* gates_s = w; w += (size_t)Bd * H8;
  float* gates_g = w; w += (size_t)Bd * H4;
  float* cg      = w; w += (size_t)Bd * Hd;
  float* bias_d  = w; w += H4;
  float* bias_s  = w; w += H8;
  float* gbias   = w; w += H4;
  float* vinit   = w; w += Hd;

  f16* h = (f16*)w;
  f16* emb16   = h; h += (size_t)Bd * Td * Ed;
  f16* dial16  = h; h += (size_t)Bd * Td * Hd;
  f16* h16     = h; h += (size_t)Bd * Hd;
  f16* x16     = h; h += (size_t)Bd * H2;
  f16* hsp16   = h; h += (size_t)Bd * H2;
  f16* spop16  = h; h += (size_t)Bd * H2;
  f16* hg16    = h; h += (size_t)Bd * Hd;
  f16* k16     = h; h += (size_t)Bd * Hd;
  f16* Wihd16  = h; h += (size_t)H4 * Ed;
  f16* Whhd16  = h; h += (size_t)H4 * Hd;
  f16* Wihs16  = h; h += (size_t)H8 * H2;
  f16* Whhs16  = h; h += (size_t)H8 * H2;
  f16* Wihg16  = h; h += (size_t)H4 * H3;
  f16* Wq16    = h; h += (size_t)Hd * Hd;
  f16* Wk16    = h; h += (size_t)Hd * Hd;
  f16* Wv16    = h; h += (size_t)Hd * Hd;

  // ---- per-call init (deterministic, graph-capturable) ----
  hipMemsetAsync(h16, 0, (size_t)Bd * Hd * sizeof(f16), stream);   // h_0 of RNN_D = 0
  hipMemsetAsync(c_d, 0, (size_t)Bd * Hd * sizeof(float), stream);
  auto cvt = [&](const float* s, f16* d, int n) {
    k_f32tof16<<<(n + 255) / 256, 256, 0, stream>>>(s, d, n);
  };
  cvt(Wih_d, Wihd16, H4 * Ed);
  cvt(Whh_d, Whhd16, H4 * Hd);
  cvt(Wih_s, Wihs16, H8 * H2);
  cvt(Whh_s, Whhs16, H8 * H2);
  cvt(Wih_g, Wihg16, H4 * H3);
  cvt(Wq, Wq16, Hd * Hd);
  cvt(Wk, Wk16, Hd * Hd);
  cvt(Wv, Wv16, Hd * Hd);
  {
    int n = Bd * Td * Ed;
    k_embed<<<(n + 255) / 256, 256, 0, stream>>>(emb_table, inputs, emb16, n);
  }
  k_addvec<<<(H4 + 255) / 256, 256, 0, stream>>>(bih_d, bhh_d, bias_d, H4);
  k_addvec<<<(H8 + 255) / 256, 256, 0, stream>>>(bih_s, bhh_s, bias_s, H8);
  k_gbias<<<(H4 + 255) / 256, 256, 0, stream>>>(Whh_g, d_h, bih_g, bhh_g, gbias);
  k_vinit<<<(Hd + 255) / 256, 256, 0, stream>>>(Wv, bk, bv, vinit);
  {
    int n = Bd * Td * Hd;
    k_initKV<<<(n + 255) / 256, 256, 0, stream>>>(Kb, Vb, bk, vinit, n);
  }
  {
    int n = Bd * NSPK * H2;
    k_initspk<<<(n + 255) / 256, 256, 0, stream>>>(spk_h, h_0, n);
  }

  // ---- phase 1: hoisted input projection (one big parallel WMMA GEMM) ----
  // xg[b,t,:] = emb[b,t,:] @ Wih_d^T + (bih_d + bhh_d)    M=3072,N=1536,K=384
  gemm(stream, emb16, Ed, Wihd16, Ed, bias_d, nullptr, 0, xg, H4, nullptr, 0,
       Bd * Td, H4, Ed);

  // serial recurrence: only h @ Whh_d^T (32x1536, K=384) per step
  for (int t = 0; t < Td; ++t) {
    gemm(stream, h16, Hd, Whhd16, Hd, nullptr, xg + (size_t)t * H4, Td * H4,
         gates_d, H4, nullptr, 0, Bd, H4, Hd);
    k_lstm_d<<<(Bd * Hd + 255) / 256, 256, 0, stream>>>(gates_d, h16, c_d, dial, dial16, t);
  }

  // ---- phase 2: 32 dialogues advance in lockstep over t ----
  for (int t = 0; t < Td; ++t) {
    // q = u @ Wq^T + bq            (u = dial[:,t,:])
    gemm(stream, dial16 + (size_t)t * Hd, Td * Hd, Wq16, Hd, bq, nullptr, 0,
         qb, Hd, nullptr, 0, Bd, Hd, Hd);
    // attention + x_s assembly + per-speaker hidden gather
    k_attn<<<Bd, 128, 0, stream>>>(qb, dial, Kb, Vb, att_h, chat_ids, spk_info,
                                   spk_h, x16, hsp16, t);
    // gates_s = x_s @ Wih_s^T + bias_s ; then += hsp @ Whh_s^T
    gemm(stream, x16, H2, Wihs16, H2, bias_s, nullptr, 0, gates_s, H8, nullptr, 0,
         Bd, H8, H2);
    gemm(stream, hsp16, H2, Whhs16, H2, nullptr, gates_s, H8, gates_s, H8, nullptr, 0,
         Bd, H8, H2);
    k_lstm_s<<<(Bd * H2 + 255) / 256, 256, 0, stream>>>(
        gates_s, dial, att_h, chat_ids, spk_info, c_s, spk_h, cg, out_spop, spop16, t);
    // gates_g = spop @ Wih_g[:, :768]^T + (d_h@Whh_g^T + biases)  [tail of fop2 is 0]
    gemm(stream, spop16, H2, Wihg16, H3, gbias, nullptr, 0, gates_g, H4, nullptr, 0,
         Bd, H4, H2);
    k_lstm_g<<<(Bd * Hd + 255) / 256, 256, 0, stream>>>(gates_g, cg, out_op, hg16, t);
    // k_new -> Kbuf[:,t,:] (f32 for attention) + k16 (f16 for v-GEMM)
    gemm(stream, hg16, Hd, Wk16, Hd, bk, nullptr, 0,
         Kb + (size_t)t * Hd, Td * Hd, k16, Hd, Bd, Hd, Hd);
    // v_new = k_new @ Wv^T + bv -> Vbuf[:,t,:]
    gemm(stream, k16, Hd, Wv16, Hd, bv, nullptr, 0,
         Vb + (size_t)t * Hd, Td * Hd, nullptr, 0, Bd, Hd, Hd);
  }
}
